// LSTMLanguageModel_4380866641936
// MI455X (gfx1250) — matrix operand order repaired and checked by reading, not compiled, not run on hardware
//
#include <hip/hip_runtime.h>
#include <stdint.h>

// ---------------------------------------------------------------------------
// Problem constants (from reference setup_inputs)
// ---------------------------------------------------------------------------
#define BB 32        // batch
#define SS 256       // sequence length
#define VV 32000     // vocab (mult of 64)
#define EE 400       // embedding dim
#define HH 1150      // hidden dim (layers 0,1); layer 2 hidden = EE
#define EP 416       // EE padded to mult of 32
#define HP 1152      // HH padded to mult of 32
#define R4H 4608     // 4*HH=4600 weight rows padded to mult of 64
#define R4E 1600     // 4*EE=1600 already mult of 64

typedef unsigned short u16;
typedef __attribute__((ext_vector_type(16))) __bf16 bf16x16;
typedef __attribute__((ext_vector_type(8)))  __bf16 bf16x8;
typedef __attribute__((ext_vector_type(8)))  float  f32x8;
typedef __attribute__((ext_vector_type(4)))  int    v4i;

// Use CDNA5 async global->LDS copies when the toolchain exposes them.
#if defined(__gfx1250__) && __has_builtin(__builtin_amdgcn_global_load_async_to_lds_b128)
#define USE_ASYNC 1
typedef __attribute__((address_space(1))) v4i* as1_v4i;
typedef __attribute__((address_space(3))) v4i* as3_v4i;
#endif

// round-to-nearest-even f32 -> bf16 bits
static __device__ __forceinline__ u16 f2bf(float f) {
    union { float f; unsigned u; } v; v.f = f;
    unsigned r = v.u + 0x7fffu + ((v.u >> 16) & 1u);
    return (u16)(r >> 16);
}

// ---------------------------------------------------------------------------
// Strided f32 -> bf16 convert: out[r*ldp + c] = bf16(in[r*K + c])
// (padding regions of `out` are pre-zeroed with hipMemsetAsync)
// ---------------------------------------------------------------------------
__global__ void k_conv_pad(const float* __restrict__ in, u16* __restrict__ out,
                           long long rows, int K, int ldp) {
    long long i = (long long)blockIdx.x * blockDim.x + threadIdx.x;
    if (i >= rows * K) return;
    long long r = i / K;
    int c = (int)(i % K);
    out[r * ldp + c] = f2bf(in[i]);
}

// ---------------------------------------------------------------------------
// Embedding gather: x[(s*B+b)*EP + e] = bf16(emb[tok[b,s], e])
// ---------------------------------------------------------------------------
__global__ void k_embed(const int* __restrict__ tok, const float* __restrict__ emb,
                        u16* __restrict__ x) {
    long long i = (long long)blockIdx.x * blockDim.x + threadIdx.x;
    const long long n = (long long)SS * BB * EE;
    if (i >= n) return;
    int e = (int)(i % EE);
    long long r = i / EE;
    int b = (int)(r % BB);
    int s = (int)(r / BB);
    int t = tok[(long long)b * SS + s];
    x[r * EP + e] = f2bf(emb[(long long)t * EE + e]);
}

// ---------------------------------------------------------------------------
// WMMA GEMM:  C[M,N] = A[M,K]*Bm[N,K]^T (+bias[N]); bf16 in, f32 out.
// 256 threads = 8 waves; block tile 64(M) x 64(N); K chunk 32; double-buffered
// LDS fed by GLOBAL_LOAD_ASYNC_TO_LDS_B128 (ASYNCcnt) when available.
// Preconditions (arranged by host-side padding/memset):
//   - lda/ldb multiples of 32; Kp multiple of 32; K-padding zero-filled
//   - A has >= ceil64(M) addressable rows; Bm has >= ceil64(N) zero-padded rows
//   - 16B alignment of all row segments
// Store remap: off(m,n) = (m % m_inner)*stride_b + (m / m_inner)*stride_s + n
// ---------------------------------------------------------------------------
#define LS 40   // LDS row stride in u16 (80B rows -> 16B-aligned segments)

__global__ __launch_bounds__(256)
void k_gemm_bf16_wmma(const u16* __restrict__ A, int lda,
                      const u16* __restrict__ Bm, int ldb,
                      const float* __restrict__ bias,
                      float* __restrict__ C,
                      int M, int N, int Kp,
                      int m_inner, long long stride_b, long long stride_s) {
    __shared__ u16 sA[2][64 * LS];
    __shared__ u16 sB[2][64 * LS];

    const int tid   = threadIdx.x;
    const int lane  = tid & 31;
    const int wv    = tid >> 5;      // 0..7
    const int half  = lane >> 4;     // 0/1
    const int l16   = lane & 15;
    const int nSub  = wv & 3;        // N subtile 0..3
    const int mPair = wv >> 2;       // M subtile pair 0/1

    const int mBase = blockIdx.y * 64;
    const int nBase = blockIdx.x * 64;

    // staging role: one 16B segment of A and one of B per thread per chunk
    const int srow = tid >> 2;           // 0..63
    const int scol = (tid & 3) * 8;      // 0,8,16,24 (u16 elems)
    const u16* gA = A  + (size_t)(mBase + srow) * lda + scol;
    const u16* gB = Bm + (size_t)(nBase + srow) * ldb + scol;
    const int ldsOff = srow * LS + scol;

    f32x8 acc0 = {};
    f32x8 acc1 = {};
    const int nk = Kp >> 5;

    auto compute = [&](int buf) {
        // B fragment: lane = col n (l16) of subtile nSub; lane<16: K {0..7,16..23}; lane>=16: K {8..15,24..31}
        const u16* bp = &sB[buf][(nSub * 16 + l16) * LS + half * 8];
        bf16x8 blo = *reinterpret_cast<const bf16x8*>(bp);
        bf16x8 bhi = *reinterpret_cast<const bf16x8*>(bp + 16);
        bf16x16 bfrag;
#pragma unroll
        for (int i = 0; i < 8; ++i) { bfrag[i] = blo[i]; bfrag[i + 8] = bhi[i]; }
        // A fragments: lane<16: M=l16, K {0..7,16..23}; lane>=16: K {8..15,24..31}
#pragma unroll
        for (int t = 0; t < 2; ++t) {
            const u16* ap = &sA[buf][((mPair * 2 + t) * 16 + l16) * LS];
            bf16x8 alo = *reinterpret_cast<const bf16x8*>(ap + half * 8);
            bf16x8 ahi = *reinterpret_cast<const bf16x8*>(ap + 16 + half * 8);
            bf16x16 afrag;
#pragma unroll
            for (int i = 0; i < 8; ++i) { afrag[i] = alo[i]; afrag[i + 8] = ahi[i]; }
            if (t == 0)
                acc0 = __builtin_amdgcn_wmma_f32_16x16x32_bf16(
                    false, afrag, false, bfrag, (short)0, acc0, false, false);
            else
                acc1 = __builtin_amdgcn_wmma_f32_16x16x32_bf16(
                    false, afrag, false, bfrag, (short)0, acc1, false, false);
        }
    };

#ifdef USE_ASYNC
    auto stage = [&](int buf, int k0) {
        __builtin_amdgcn_global_load_async_to_lds_b128(
            (as1_v4i)(gA + k0), (as3_v4i)(&sA[buf][ldsOff]), 0, 0);
        __builtin_amdgcn_global_load_async_to_lds_b128(
            (as1_v4i)(gB + k0), (as3_v4i)(&sB[buf][ldsOff]), 0, 0);
    };
    stage(0, 0);
    for (int kc = 0; kc < nk; ++kc) {
        const int cur = kc & 1;
#if __has_builtin(__builtin_amdgcn_s_wait_asynccnt)
        __builtin_amdgcn_s_wait_asynccnt(0);     // my copies for `cur` landed
#else
        asm volatile("s_wait_asynccnt 0" ::: "memory");
#endif
        __syncthreads();                         // everyone's copies landed; prev compute done
        if (kc + 1 < nk) stage(cur ^ 1, (kc + 1) * 32);   // prefetch next chunk
        compute(cur);
    }
#else
    // Fallback: vectorized global_load_b128 -> register prefetch -> ds_store_b128
    uint4 ra, rb;
    ra = *reinterpret_cast<const uint4*>(gA);
    rb = *reinterpret_cast<const uint4*>(gB);
    for (int kc = 0; kc < nk; ++kc) {
        *reinterpret_cast<uint4*>(&sA[0][ldsOff]) = ra;
        *reinterpret_cast<uint4*>(&sB[0][ldsOff]) = rb;
        __syncthreads();
        if (kc + 1 < nk) {
            ra = *reinterpret_cast<const uint4*>(gA + (kc + 1) * 32);
            rb = *reinterpret_cast<const uint4*>(gB + (kc + 1) * 32);
        }
        compute(0);
        __syncthreads();
    }
#endif

    // store: C/D 16x16 f32 layout: VGPR r -> M = r + 8*half, N = l16
    const int nG = nBase + nSub * 16 + l16;
    if (nG < N) {
        const float bv = bias ? bias[nG] : 0.0f;
#pragma unroll
        for (int t = 0; t < 2; ++t) {
            const f32x8& acc = t ? acc1 : acc0;
#pragma unroll
            for (int r = 0; r < 8; ++r) {
                int mG = mBase + (mPair * 2 + t) * 16 + r + 8 * half;
                if (mG < M) {
                    long long off = (long long)(mG % m_inner) * stride_b +
                                    (long long)(mG / m_inner) * stride_s + nG;
                    C[off] = acc[r] + bv;
                }
            }
        }
    }
}

// ---------------------------------------------------------------------------
// LSTM pointwise: gates = xw[s] + g(recurrent); update c; emit h (bf16, padded)
// gate order i,f,g,o in chunks of Hl (PyTorch order).
// ---------------------------------------------------------------------------
__global__ void k_lstm_point(const float* __restrict__ xw,   // [B,4H] slab, step s
                             const float* __restrict__ g,    // [B,4H] recurrent
                             float* __restrict__ c,          // [B,H] fp32 cell
                             u16* __restrict__ hout,         // [B, rowp] bf16
                             int Hl, int rowp) {
    int idx = blockIdx.x * blockDim.x + threadIdx.x;
    if (idx >= BB * Hl) return;
    int b = idx / Hl, j = idx % Hl;
    long long base = (long long)b * 4 * Hl;
    float gi = xw[base + j]          + g[base + j];
    float gf = xw[base + Hl + j]     + g[base + Hl + j];
    float gg = xw[base + 2 * Hl + j] + g[base + 2 * Hl + j];
    float go = xw[base + 3 * Hl + j] + g[base + 3 * Hl + j];
    float i_ = 1.0f / (1.0f + __expf(-gi));
    float f_ = 1.0f / (1.0f + __expf(-gf));
    float t_ = tanhf(gg);
    float o_ = 1.0f / (1.0f + __expf(-go));
    float cn = f_ * c[idx] + i_ * t_;
    c[idx] = cn;
    float h = o_ * tanhf(cn);
    hout[(long long)b * rowp + j] = f2bf(h);
}

// ---------------------------------------------------------------------------
// Host-side orchestration
// ---------------------------------------------------------------------------
static inline size_t align256(size_t x) { return (x + 255) & ~(size_t)255; }

extern "C" void kernel_launch(void* const* d_in, const int* in_sizes, int n_in,
                              void* d_out, int out_size, void* d_ws, size_t ws_size,
                              hipStream_t stream) {
    (void)in_sizes; (void)n_in; (void)out_size; (void)ws_size;
    const int*   tok   = (const int*)  d_in[0];
    const float* emb   = (const float*)d_in[1];
    const float* Wih0  = (const float*)d_in[2];
    const float* Whh0  = (const float*)d_in[3];
    const float* b0    = (const float*)d_in[4];
    const float* Wih1  = (const float*)d_in[5];
    const float* Whh1  = (const float*)d_in[6];
    const float* b1    = (const float*)d_in[7];
    const float* Wih2  = (const float*)d_in[8];
    const float* Whh2  = (const float*)d_in[9];
    const float* b2    = (const float*)d_in[10];
    const float* dec_b = (const float*)d_in[11];
    float* out = (float*)d_out;

    // ---- workspace carve-up (all bf16 buffers padded: lda%32==0, rows%64==0) ----
    uint8_t* ws = (uint8_t*)d_ws;
    size_t off = 0;
    auto take = [&](size_t bytes) -> uint8_t* {
        uint8_t* p = ws + off; off = align256(off + bytes); return p;
    };
    struct Buf { u16* p; size_t bytes; };
    auto takeB = [&](size_t elems) -> Buf { Buf b; b.bytes = elems * 2; b.p = (u16*)take(b.bytes); return b; };

    Buf emb_bf = takeB((size_t)VV * EP);
    Buf wih0   = takeB((size_t)R4H * EP);
    Buf whh0   = takeB((size_t)R4H * HP);
    Buf wih1   = takeB((size_t)R4H * HP);
    Buf whh1   = takeB((size_t)R4H * HP);
    Buf wih2   = takeB((size_t)R4E * HP);
    Buf whh2   = takeB((size_t)R4E * EP);
    Buf x_bf   = takeB((size_t)SS * BB * EP);
    Buf hs1    = takeB((size_t)(SS + 1) * BB * HP);   // row 0 = h(-1) = 0
    Buf hs2    = takeB((size_t)(SS + 1) * BB * HP);
    Buf hs3    = takeB((size_t)(SS + 1) * BB * EP);
    float* xw   = (float*)take((size_t)SS * BB * 4 * HH * 4);   // reused per layer
    float* gbuf = (float*)take((size_t)BB * 4 * HH * 4);
    float* cbuf = (float*)take((size_t)BB * HH * 4);

    // ---- 0. zero-fill padded bf16 buffers (padding must be exactly 0) ----
    const Buf* zb[] = { &emb_bf, &wih0, &whh0, &wih1, &whh1, &wih2, &whh2,
                        &x_bf, &hs1, &hs2, &hs3 };
    for (auto* b : zb) (void)hipMemsetAsync(b->p, 0, b->bytes, stream);

    // ---- 1. weight / table conversion to padded bf16 ----
    auto conv = [&](const float* src, u16* dst, long long rows, int K, int ldp) {
        long long n = rows * K;
        hipLaunchKernelGGL(k_conv_pad, dim3((unsigned)((n + 255) / 256)), dim3(256), 0, stream,
                           src, dst, rows, K, ldp);
    };
    conv(emb,  emb_bf.p, VV,      EE, EP);
    conv(Wih0, wih0.p,   4 * HH,  EE, EP);
    conv(Whh0, whh0.p,   4 * HH,  HH, HP);
    conv(Wih1, wih1.p,   4 * HH,  HH, HP);
    conv(Whh1, whh1.p,   4 * HH,  HH, HP);
    conv(Wih2, wih2.p,   4 * EE,  HH, HP);
    conv(Whh2, whh2.p,   4 * EE,  EE, EP);

    // ---- 2. embedding gather -> [S*B, EP] bf16 ----
    {
        long long n = (long long)SS * BB * EE;
        hipLaunchKernelGGL(k_embed, dim3((unsigned)((n + 255) / 256)), dim3(256), 0, stream,
                           tok, emb, x_bf.p);
    }

    auto gemm = [&](const u16* A, int lda, const u16* Bm, int ldb, const float* bias,
                    float* C, int M, int N, int Kp,
                    int m_inner, long long sb, long long sstep) {
        dim3 grid((N + 63) / 64, (M + 63) / 64);
        hipLaunchKernelGGL(k_gemm_bf16_wmma, grid, dim3(256), 0, stream,
                           A, lda, Bm, ldb, bias, C, M, N, Kp, m_inner, sb, sstep);
    };
    const int BIGM = 1 << 30;

    // ---- 3. three LSTM layers ----
    struct Cfg {
        const u16* A; int lda;           // layer input (S*B rows)
        const u16* Wih; int Kin;         // padded K of input GEMM (== lda)
        const u16* Whh; int Krec;        // padded K of recurrent GEMM
        const float* bias; int Hl;       // true hidden size
        u16* hs; int rowp;               // hidden state buffer, padded row len
    };
    const Cfg cfg[3] = {
        { x_bf.p,                  EP, wih0.p, EP, whh0.p, HP, b0, HH, hs1.p, HP },
        { hs1.p + (size_t)BB * HP, HP, wih1.p, HP, whh1.p, HP, b1, HH, hs2.p, HP },
        { hs2.p + (size_t)BB * HP, HP, wih2.p, HP, whh2.p, EP, b2, EE, hs3.p, EP },
    };

    for (int l = 0; l < 3; ++l) {
        const Cfg& c = cfg[l];
        const int H4 = 4 * c.Hl;
        // xW = X @ Wih^T + b over all S*B rows at once
        gemm(c.A, c.lda, c.Wih, c.Kin, c.bias, xw,
             SS * BB, H4, c.Kin, BIGM, (long long)H4, 0);
        (void)hipMemsetAsync(cbuf, 0, (size_t)BB * c.Hl * 4, stream);   // cell = 0
        const int ptThr = 256;
        const int ptBlk = (BB * c.Hl + ptThr - 1) / ptThr;
        for (int s = 0; s < SS; ++s) {
            // gbuf = h(s-1) @ Whh^T   (M=32; extra tile rows read in-bounds garbage,
            // discarded by the guarded store)
            gemm(c.hs + (size_t)s * BB * c.rowp, c.rowp, c.Whh, c.Krec, nullptr, gbuf,
                 BB, H4, c.Krec, BIGM, (long long)H4, 0);
            hipLaunchKernelGGL(k_lstm_point, dim3(ptBlk), dim3(ptThr), 0, stream,
                               xw + (size_t)s * BB * H4, gbuf, cbuf,
                               c.hs + (size_t)(s + 1) * BB * c.rowp, c.Hl, c.rowp);
        }
    }

    // ---- 4. tied decoder: out[b,s,:] = h3[s,b,:] @ emb^T + dec_b ----
    // rows m = s*B + b stored directly into [B,S,V] via remap (no transpose pass)
    gemm(hs3.p + (size_t)BB * EP, EP, emb_bf.p, EP, dec_b, out,
         SS * BB, VV, EP,
         /*m_inner=*/BB, /*stride_b=*/(long long)SS * VV, /*stride_s=*/(long long)VV);
}
